// DiffusionActiveInference_33071248179483
// MI455X (gfx1250) — compile-verified, matrix operand order checked by reading, not run on hardware
//
#include <hip/hip_runtime.h>
#include <cstdint>
#include <cstddef>

// ---------------------------------------------------------------------------
// MI455X (gfx1250) implementation of the DiffusionActiveInference EFE rollout.
// All matmuls run through v_wmma_f32_16x16x32_f16 (f16 in, f32 accum).
// ---------------------------------------------------------------------------

typedef _Float16 h16;
typedef __attribute__((ext_vector_type(16))) _Float16 v16h;
typedef __attribute__((ext_vector_type(8)))  _Float16 v8h;
typedef __attribute__((ext_vector_type(8)))  float    v8f;

#define BREF     512
#define TRAJ     15
#define NROWS    (BREF * TRAJ)   // 7680, multiple of 32
#define HORIZON  5
#define NSAMP    10
#define TWO_PI_E 17.0794684453f
#define LOG_2PIE 2.8378770664093453f

// ----------------------------- RNG helpers ---------------------------------
__device__ __forceinline__ uint32_t hashu(uint32_t a, uint32_t b, uint32_t c) {
  uint32_t h = a * 0x9E3779B1u + 0x7F4A7C15u;
  h ^= b * 0x85EBCA77u; h = (h ^ (h >> 16)) * 0xC2B2AE3Du;
  h ^= c * 0x27D4EB2Fu; h = (h ^ (h >> 15)) * 0x165667B1u;
  h ^= h >> 13;
  return h;
}
__device__ __forceinline__ float rng_uniform(uint32_t a, uint32_t b, uint32_t c) {
  return (float)(hashu(a, b, c) >> 8) * (1.0f / 16777216.0f);
}
__device__ __forceinline__ float rng_normal(uint32_t a, uint32_t b, uint32_t c) {
  uint32_t h1 = hashu(a, b, c);
  uint32_t h2 = hashu(h1 ^ 0x6C62272Eu, b + 0x9E3779B9u, c ^ 0x5851F42Du);
  float u1 = (float)((h1 >> 8) + 1u) * (1.0f / 16777216.0f);   // (0,1]
  float u2 = (float)(h2 >> 8) * (1.0f / 16777216.0f);
  return sqrtf(-2.0f * __logf(u1)) * __cosf(6.28318530718f * u2);
}

// -------------------- weight repack: f32 [K,N] -> WMMA B frags -------------
// Fragment layout (16-bit B matrix, 32x16 K x N block):
//   lane<16 : col = lane,    K = {i..} 0..7 then 16..23 over i=0..15
//   lane>=16: col = lane-16, K = 8..15 then 24..31
// dst index = ((kb*(Np/16)+nb)*32 + lane)*16 + i
__global__ void k_pack(const float* __restrict__ W, h16* __restrict__ dst,
                       int K, int N, int Kp, int Np, size_t total) {
  size_t idx = (size_t)blockIdx.x * blockDim.x + threadIdx.x;
  if (idx >= total) return;
  int i    = (int)(idx & 15);
  int lane = (int)((idx >> 4) & 31);
  size_t rest = idx >> 9;
  int NB16 = Np >> 4;
  int nb = (int)(rest % (size_t)NB16);
  int kb = (int)(rest / (size_t)NB16);
  int kl = (lane < 16) ? ((i < 8) ? i : i + 8) : ((i < 8) ? i + 8 : i + 16);
  int k  = kb * 32 + kl;
  int nn = nb * 16 + (lane & 15);
  float v = (k < K && nn < N) ? W[(size_t)k * N + nn] : 0.0f;
  dst[idx] = (h16)v;
}

// ------------------------------- WMMA GEMM ---------------------------------
// C[M,Np] = A[M,Kp](f16, row-major) @ Wpacked + bias.  4 waves/block, each
// wave computes a 32x32 tile with 4 accumulators (2 A-frags x 2 B-frags).
__global__ void k_gemm(const h16* __restrict__ A, const h16* __restrict__ Bp,
                       const float* __restrict__ bias, float* __restrict__ C,
                       int M, int Kp, int Np, int Nreal) {
  const int lane = threadIdx.x & 31;
  const int MT = M >> 5, NT = Np >> 5;
  int wave = blockIdx.x * 4 + (threadIdx.x >> 5);
  if (wave >= MT * NT) return;                 // wave-uniform
  const int mt = wave % MT, nt = wave / MT;
  const int mBase = mt << 5, nBase = nt << 5;
  const int l16 = lane & 15, hi = lane >> 4;
  const int NB16 = Np >> 4;
  const int nb16 = nBase >> 4;

  const h16* Ar0 = A + (size_t)(mBase + l16) * Kp + hi * 8;
  const h16* Ar1 = Ar0 + (size_t)16 * Kp;
  const h16* Bq  = Bp + ((size_t)nb16 * 32 + lane) * 16;
  const size_t bStride = (size_t)NB16 * 32 * 16;   // halves per kb step

  v8f c00 = {}, c01 = {}, c10 = {}, c11 = {};
  const int KB = Kp >> 5;
  for (int kb = 0; kb < KB; ++kb) {
    union { v16h v; v8h h[2]; } a0, a1;
    a0.h[0] = *reinterpret_cast<const v8h*>(Ar0 + kb * 32);
    a0.h[1] = *reinterpret_cast<const v8h*>(Ar0 + kb * 32 + 16);
    a1.h[0] = *reinterpret_cast<const v8h*>(Ar1 + kb * 32);
    a1.h[1] = *reinterpret_cast<const v8h*>(Ar1 + kb * 32 + 16);
    v16h b0 = *reinterpret_cast<const v16h*>(Bq + (size_t)kb * bStride);
    v16h b1 = *reinterpret_cast<const v16h*>(Bq + (size_t)kb * bStride + 512);
    c00 = __builtin_amdgcn_wmma_f32_16x16x32_f16(false, a0.v, false, b0, (short)0, c00, false, false);
    c01 = __builtin_amdgcn_wmma_f32_16x16x32_f16(false, a0.v, false, b1, (short)0, c01, false, false);
    c10 = __builtin_amdgcn_wmma_f32_16x16x32_f16(false, a1.v, false, b0, (short)0, c10, false, false);
    c11 = __builtin_amdgcn_wmma_f32_16x16x32_f16(false, a1.v, false, b1, (short)0, c11, false, false);
  }
  // Epilogue: accumulator VGPR r holds (M=r, N=l16) for lane<16, (M=r+8) else.
  int rowBase = mBase + hi * 8;
  int col0 = nBase + l16, col1 = col0 + 16;
  float bb0 = (col0 < Nreal) ? bias[col0] : 0.0f;
  float bb1 = (col1 < Nreal) ? bias[col1] : 0.0f;
  #pragma unroll
  for (int r = 0; r < 8; ++r) {
    C[(size_t)(rowBase + r)      * Np + col0] = c00[r] + bb0;
    C[(size_t)(rowBase + r)      * Np + col1] = c01[r] + bb1;
    C[(size_t)(rowBase + 16 + r) * Np + col0] = c10[r] + bb0;
    C[(size_t)(rowBase + 16 + r) * Np + col1] = c11[r] + bb1;
  }
}

// --------------------------- elementwise kernels ---------------------------
__global__ void k_zero(float* p, size_t total) {
  size_t i = (size_t)blockIdx.x * blockDim.x + threadIdx.x;
  if (i < total) p[i] = 0.0f;
}

__global__ void k_tile_latent(const float* __restrict__ latent, h16* __restrict__ z,
                              size_t total /* NROWS*128 */) {
  size_t i = (size_t)blockIdx.x * blockDim.x + threadIdx.x;
  if (i >= total) return;
  size_t row = i >> 7;              // 0..NROWS-1
  size_t b   = row % BREF;          // tiled source row
  z[i] = (h16)latent[(b << 7) + (i & 127)];
}

__global__ void k_relu_f16(const float* __restrict__ in, int stride,
                           h16* __restrict__ out, int C, size_t total) {
  size_t i = (size_t)blockIdx.x * blockDim.x + threadIdx.x;
  if (i >= total) return;
  size_t row = i / (size_t)C;
  int c = (int)(i % (size_t)C);
  out[i] = (h16)fmaxf(in[row * stride + c], 0.0f);
}

__global__ void k_clipf(float* p, float lo, float hi, size_t total) {
  size_t i = (size_t)blockIdx.x * blockDim.x + threadIdx.x;
  if (i >= total) return;
  p[i] = fminf(fmaxf(p[i], lo), hi);
}

// LayerNorm -> (SiLU|ReLU) -> optional dropout -> optional residual, f16 out
__global__ void k_ln_act(const float* __restrict__ in, int stride,
                         const float* __restrict__ g, const float* __restrict__ be,
                         const h16* __restrict__ res, h16* __restrict__ out,
                         int C, int silu, int drop,
                         const int* __restrict__ seedp, uint32_t tag) {
  __shared__ float s1[256], s2[256];
  int row = blockIdx.x, tid = threadIdx.x;
  const float* x = in + (size_t)row * stride;
  float a = 0.0f, b = 0.0f;
  for (int c = tid; c < C; c += 256) { float v = x[c]; a += v; b += v * v; }
  s1[tid] = a; s2[tid] = b; __syncthreads();
  for (int off = 128; off > 0; off >>= 1) {
    if (tid < off) { s1[tid] += s1[tid + off]; s2[tid] += s2[tid + off]; }
    __syncthreads();
  }
  float mean = s1[0] / (float)C;
  float var  = s2[0] / (float)C - mean * mean;
  float rs = rsqrtf(var + 1e-5f);
  uint32_t seed = (uint32_t)(*seedp);
  for (int c = tid; c < C; c += 256) {
    float y = (x[c] - mean) * rs * g[c] + be[c];
    y = silu ? (y / (1.0f + __expf(-y))) : fmaxf(y, 0.0f);
    if (drop) {
      float u = rng_uniform(seed, tag, (uint32_t)(row * C + c));
      y = (u < 0.2f) ? 0.0f : y * 1.25f;
    }
    if (res) y += (float)res[(size_t)row * C + c];
    out[(size_t)row * C + c] = (h16)y;
  }
}

// za[n,160] = [z(128) | action(16) | pad(16)], action = mu + exp(ls)*eps
__global__ void k_action_concat(const h16* __restrict__ zf,
                                const float* __restrict__ muC,
                                const float* __restrict__ lsC,
                                const int* __restrict__ seedp, uint32_t tag,
                                h16* __restrict__ za, size_t total) {
  size_t i = (size_t)blockIdx.x * blockDim.x + threadIdx.x;
  if (i >= total) return;
  int col = (int)(i % 160);
  size_t row = i / 160;
  h16 v;
  if (col < 128) v = zf[row * 128 + col];
  else if (col < 144) {
    int j = col - 128;
    float mu = muC[row * 32 + j];
    float ls = lsC[row * 32 + j];      // pre-clipped [-5,2]
    float eps = rng_normal((uint32_t)(*seedp), tag, (uint32_t)(row * 16 + j));
    v = (h16)(mu + __expf(ls) * eps);
  } else v = (h16)0.0f;
  za[i] = v;
}

__global__ void k_entropy_pol(const float* __restrict__ lsC, float* __restrict__ ent, int n) {
  int i = blockIdx.x * blockDim.x + threadIdx.x;
  if (i >= n) return;
  float s = 0.0f;
  for (int j = 0; j < 16; ++j) s += 0.5f * LOG_2PIE + lsC[(size_t)i * 32 + j];
  ent[i] = s;
}

// out[n,128] f16 = mean + exp(0.5*lv)*eps
__global__ void k_sample_f16(const float* __restrict__ mean, const float* __restrict__ lv,
                             const int* __restrict__ seedp, uint32_t tag,
                             h16* __restrict__ out, size_t total) {
  size_t i = (size_t)blockIdx.x * blockDim.x + threadIdx.x;
  if (i >= total) return;
  float eps = rng_normal((uint32_t)(*seedp), tag, (uint32_t)i);
  out[i] = (h16)(mean[i] + __expf(0.5f * lv[i]) * eps);
}

// Vin[n,256] = [nz(128) | time-embed(128)]
__global__ void k_concat_value(const h16* __restrict__ nz, float t,
                               h16* __restrict__ out, size_t total) {
  size_t i = (size_t)blockIdx.x * blockDim.x + threadIdx.x;
  if (i >= total) return;
  int col = (int)(i % 256);
  size_t row = i / 256;
  if (col < 128) out[i] = nz[row * 128 + col];
  else {
    int j = col - 128;
    int jj = (j < 64) ? j : j - 64;
    float fr = __expf(-9.21034037198f * (float)jj * (1.0f / 64.0f)); // ln(1e4)
    float ang = t * fr;
    out[i] = (h16)((j < 64) ? __sinf(ang) : __cosf(ang));
  }
}

__global__ void k_acc(const float* __restrict__ obs, float* __restrict__ sum,
                      float* __restrict__ sq, size_t total) {
  size_t i = (size_t)blockIdx.x * blockDim.x + threadIdx.x;
  if (i >= total) return;
  float v = obs[i];
  sum[i] += v; sq[i] += v * v;
}

// per-row: h = sum_c 0.5*log(2*pi*e*var + addC), var unbiased over NSAMP
__global__ void k_var_entropy(const float* __restrict__ sum, const float* __restrict__ sq,
                              float* __restrict__ h, float addC) {
  __shared__ float red[256];
  int row = blockIdx.x, tid = threadIdx.x;
  float s = sum[(size_t)row * 256 + tid], q = sq[(size_t)row * 256 + tid];
  float var = (q - s * s * (1.0f / NSAMP)) * (1.0f / (NSAMP - 1));
  var = fmaxf(var, 0.0f);
  red[tid] = 0.5f * __logf(TWO_PI_E * var + addC);
  __syncthreads();
  for (int off = 128; off > 0; off >>= 1) {
    if (tid < off) red[tid] += red[tid + off];
    __syncthreads();
  }
  if (tid == 0) h[row] = red[0];
}

__global__ void k_combine(float* __restrict__ traj, const float* __restrict__ hp,
                          const float* __restrict__ ht, const float* __restrict__ rC,
                          const float* __restrict__ vC, const float* __restrict__ ent,
                          float gpow, int n) {
  int i = blockIdx.x * blockDim.x + threadIdx.x;
  if (i >= n) return;
  float epi  = fmaxf(hp[i] - ht[i], 0.0f);
  float prag = rC[(size_t)i * 32] + vC[(size_t)i * 32];   // PRAG_W=1, PREF_T=1
  traj[i] += gpow * (epi + prag - 0.1f * ent[i]);         // CONS_W=0.1, -entropy
}

__global__ void k_final(const float* __restrict__ traj, float* __restrict__ out) {
  int b = blockIdx.x * blockDim.x + threadIdx.x;
  if (b >= BREF) return;
  float s = 0.0f;
  for (int t = 0; t < TRAJ; ++t) s += traj[(size_t)t * BREF + b];
  out[b] = s * (1.0f / TRAJ);
}

// ---------------------------------------------------------------------------
// Host orchestration.
// d_in order (jax tree: dict keys sorted; capitals before lowercase):
//  0: latent [512,128]
//  dec:    1:W0[128,512] 2:W1[512,512] 3:W2[512,256] 4:W3[256,256]
//          5:b0 6:b1 7:b2 8:b3 9:be0 10:be1 11:be2 12:g0 13:g1 14:g2
//  dyn:    15:W1[144,256] 16:W2[256,256] 17:Wm[256,128] 18:Wv[256,128]
//          19:b1 20:b2 21:bm 22:bv
//  policy: 23:W1[128,256] 24:W2[256,256] 25:Wls[256,16] 26:Wmu[256,16]
//          27:b1 28:b2 29:bls 30:bmu
//  reward: 31:W1[128,256] 32:W2[256,128] 33:W3[128,2] 34:b1 35:b2 36:b3
//          37:be1 38:g1
//  value:  39:W1[256,256] 40:W2[256,256] 41:W3[256,1] 42:b1 43:b2 44:b3
//  45:seed 46:horizon 47:num_trajectories 48:num_ambiguity_samples
// ---------------------------------------------------------------------------
static inline int rup32(int x) { return (x + 31) & ~31; }

extern "C" void kernel_launch(void* const* d_in, const int* in_sizes, int n_in,
                              void* d_out, int out_size, void* d_ws, size_t ws_size,
                              hipStream_t stream) {
  (void)in_sizes; (void)n_in; (void)out_size; (void)ws_size;
  char* ws = (char*)d_ws;
  size_t off = 0;
  auto alloc = [&](size_t bytes) {
    size_t o = off;
    off = (off + bytes + 255) & ~(size_t)255;
    return o;
  };
  const int* seedp = (const int*)d_in[45];

  auto launchN = [&](size_t total) { return (unsigned)((total + 255) / 256); };

  // ---- repack all weights to f16 B-fragment layout -------------------------
  auto packW = [&](int idx, int K, int N) {
    int Kp = rup32(K), Np = rup32(N);
    size_t total = (size_t)(Kp / 32) * (size_t)(Np / 16) * 512;
    size_t o = alloc(total * sizeof(h16));
    k_pack<<<launchN(total), 256, 0, stream>>>((const float*)d_in[idx],
                                               (h16*)(ws + o), K, N, Kp, Np, total);
    return o;
  };
  size_t pk_decW0 = packW(1, 128, 512);
  size_t pk_decW1 = packW(2, 512, 512);
  size_t pk_decW2 = packW(3, 512, 256);
  size_t pk_decW3 = packW(4, 256, 256);
  size_t pk_dynW1 = packW(15, 144, 256);
  size_t pk_dynW2 = packW(16, 256, 256);
  size_t pk_dynWm = packW(17, 256, 128);
  size_t pk_dynWv = packW(18, 256, 128);
  size_t pk_polW1 = packW(23, 128, 256);
  size_t pk_polW2 = packW(24, 256, 256);
  size_t pk_polWls = packW(25, 256, 16);
  size_t pk_polWmu = packW(26, 256, 16);
  size_t pk_rW1 = packW(31, 128, 256);
  size_t pk_rW2 = packW(32, 256, 128);
  size_t pk_rW3 = packW(33, 128, 2);
  size_t pk_vW1 = packW(39, 256, 256);
  size_t pk_vW2 = packW(40, 256, 256);
  size_t pk_vW3 = packW(41, 256, 1);

  // ---- activation / scratch buffers ---------------------------------------
  const size_t n = NROWS;
  size_t o_zA  = alloc(n * 128 * sizeof(h16));
  size_t o_zB  = alloc(n * 128 * sizeof(h16));
  size_t o_G1  = alloc(n * 512 * sizeof(float));
  size_t o_G2  = alloc(n * 512 * sizeof(float));
  size_t o_F1  = alloc(n * 512 * sizeof(h16));
  size_t o_F2  = alloc(n * 512 * sizeof(h16));
  size_t o_F3  = alloc(n * 512 * sizeof(h16));
  size_t o_za  = alloc(n * 160 * sizeof(h16));
  size_t o_Vin = alloc(n * 256 * sizeof(h16));
  size_t o_Din = alloc(n * 128 * sizeof(h16));
  size_t o_mu  = alloc(n * 32 * sizeof(float));
  size_t o_ls  = alloc(n * 32 * sizeof(float));
  size_t o_rC  = alloc(n * 32 * sizeof(float));
  size_t o_vC  = alloc(n * 32 * sizeof(float));
  size_t o_nm  = alloc(n * 128 * sizeof(float));
  size_t o_nlv = alloc(n * 128 * sizeof(float));
  size_t o_sum = alloc(n * 256 * sizeof(float));
  size_t o_sq  = alloc(n * 256 * sizeof(float));
  size_t o_ent = alloc(n * sizeof(float));
  size_t o_hp  = alloc(n * sizeof(float));
  size_t o_ht  = alloc(n * sizeof(float));
  size_t o_tr  = alloc(n * sizeof(float));

  auto gemm = [&](size_t aOff, size_t bOff, int biasIdx, size_t cOff,
                  int Kp, int Np, int Nreal) {
    int waves = (NROWS / 32) * (Np / 32);
    k_gemm<<<dim3((waves + 3) / 4), dim3(128), 0, stream>>>(
        (const h16*)(ws + aOff), (const h16*)(ws + bOff),
        (const float*)d_in[biasIdx], (float*)(ws + cOff),
        NROWS, Kp, Np, Nreal);
  };
  auto relu = [&](size_t inOff, int stride, size_t outOff, int C) {
    size_t total = n * (size_t)C;
    k_relu_f16<<<launchN(total), 256, 0, stream>>>((const float*)(ws + inOff),
                                                   stride, (h16*)(ws + outOff), C, total);
  };
  auto lnact = [&](size_t inOff, int stride, int gIdx, int beIdx, size_t resOff,
                   bool hasRes, size_t outOff, int C, int silu, int drop, uint32_t tag) {
    k_ln_act<<<dim3((unsigned)n), dim3(256), 0, stream>>>(
        (const float*)(ws + inOff), stride, (const float*)d_in[gIdx],
        (const float*)d_in[beIdx], hasRes ? (const h16*)(ws + resOff) : (const h16*)nullptr,
        (h16*)(ws + outOff), C, silu, drop, seedp, tag);
  };
  auto zero = [&](size_t oOff, size_t cnt) {
    k_zero<<<launchN(cnt), 256, 0, stream>>>((float*)(ws + oOff), cnt);
  };

  // ---- init: z = tile(latent), traj_efe = 0 -------------------------------
  {
    size_t total = n * 128;
    k_tile_latent<<<launchN(total), 256, 0, stream>>>((const float*)d_in[0],
                                                      (h16*)(ws + o_zA), total);
  }
  zero(o_tr, n);

  size_t zCur = o_zA, zNext = o_zB;
  float gpow = 1.0f;
  for (int t = 0; t < HORIZON; ++t) {
    uint32_t tb = (uint32_t)t;
    // ---- policy -----------------------------------------------------------
    gemm(zCur, pk_polW1, 27, o_G1, 128, 256, 256);
    relu(o_G1, 256, o_F1, 256);
    gemm(o_F1, pk_polW2, 28, o_G2, 256, 256, 256);
    relu(o_G2, 256, o_F3, 256);
    gemm(o_F3, pk_polWmu, 30, o_mu, 256, 32, 16);
    gemm(o_F3, pk_polWls, 29, o_ls, 256, 32, 16);
    k_clipf<<<launchN(n * 32), 256, 0, stream>>>((float*)(ws + o_ls), -5.0f, 2.0f, n * 32);
    k_entropy_pol<<<launchN(n), 256, 0, stream>>>((const float*)(ws + o_ls),
                                                  (float*)(ws + o_ent), (int)n);
    k_action_concat<<<launchN(n * 160), 256, 0, stream>>>(
        (const h16*)(ws + zCur), (const float*)(ws + o_mu), (const float*)(ws + o_ls),
        seedp, 0x1000u + tb, (h16*)(ws + o_za), n * 160);
    // ---- latent dynamics --------------------------------------------------
    gemm(o_za, pk_dynW1, 19, o_G1, 160, 256, 256);
    relu(o_G1, 256, o_F1, 256);
    gemm(o_F1, pk_dynW2, 20, o_G2, 256, 256, 256);
    relu(o_G2, 256, o_F3, 256);
    gemm(o_F3, pk_dynWm, 21, o_nm, 256, 128, 128);
    gemm(o_F3, pk_dynWv, 22, o_nlv, 256, 128, 128);
    k_clipf<<<launchN(n * 128), 256, 0, stream>>>((float*)(ws + o_nlv), -6.0f, 2.0f, n * 128);
    k_sample_f16<<<launchN(n * 128), 256, 0, stream>>>(
        (const float*)(ws + o_nm), (const float*)(ws + o_nlv), seedp, 0x2000u + tb,
        (h16*)(ws + zNext), n * 128);
    // ---- reward predictor -------------------------------------------------
    gemm(zNext, pk_rW1, 34, o_G1, 128, 256, 256);
    lnact(o_G1, 256, 38, 37, 0, false, o_F1, 256, /*silu=*/0, /*drop=*/0, 0u);
    gemm(o_F1, pk_rW2, 35, o_G2, 256, 128, 128);
    relu(o_G2, 128, o_F3, 128);
    gemm(o_F3, pk_rW3, 36, o_rC, 128, 32, 2);
    // ---- value network ----------------------------------------------------
    k_concat_value<<<launchN(n * 256), 256, 0, stream>>>(
        (const h16*)(ws + zNext), (float)t, (h16*)(ws + o_Vin), n * 256);
    gemm(o_Vin, pk_vW1, 42, o_G1, 256, 256, 256);
    relu(o_G1, 256, o_F1, 256);
    gemm(o_F1, pk_vW2, 43, o_G2, 256, 256, 256);
    relu(o_G2, 256, o_F3, 256);
    gemm(o_F3, pk_vW3, 44, o_vC, 256, 32, 1);
    // ---- epistemic term: 2 passes x NSAMP decoder runs --------------------
    for (int pass = 0; pass < 2; ++pass) {       // 0 = train (dropout), 1 = eval
      zero(o_sum, n * 256);
      zero(o_sq, n * 256);
      for (int s = 0; s < NSAMP; ++s) {
        uint32_t stag = ((pass == 0) ? 0x3000u : 0x4000u) + tb * 16u + (uint32_t)s;
        k_sample_f16<<<launchN(n * 128), 256, 0, stream>>>(
            (const float*)(ws + o_nm), (const float*)(ws + o_nlv), seedp, stag,
            (h16*)(ws + o_Din), n * 128);
        uint32_t dtag = 0x5000u + (tb * NSAMP + (uint32_t)s) * 4u;
        int drop = (pass == 0) ? 1 : 0;
        gemm(o_Din, pk_decW0, 5, o_G1, 128, 512, 512);
        lnact(o_G1, 512, 12, 9, 0, false, o_F1, 512, 1, drop, dtag + 0);
        gemm(o_F1, pk_decW1, 6, o_G2, 512, 512, 512);
        lnact(o_G2, 512, 13, 10, o_F1, true, o_F2, 512, 1, drop, dtag + 1);  // +h1
        gemm(o_F2, pk_decW2, 7, o_G1, 512, 256, 256);
        lnact(o_G1, 256, 14, 11, 0, false, o_F3, 256, 1, drop, dtag + 2);
        gemm(o_F3, pk_decW3, 8, o_G2, 256, 256, 256);
        k_acc<<<launchN(n * 256), 256, 0, stream>>>(
            (const float*)(ws + o_G2), (float*)(ws + o_sum), (float*)(ws + o_sq), n * 256);
      }
      // train: log(2pi*e*(var+1e-8)+1e-8); eval: log(2pi*e*var+1e-8)
      float addC = (pass == 0) ? (TWO_PI_E * 1e-8f + 1e-8f) : 1e-8f;
      k_var_entropy<<<dim3((unsigned)n), dim3(256), 0, stream>>>(
          (const float*)(ws + o_sum), (const float*)(ws + o_sq),
          (float*)(ws + ((pass == 0) ? o_hp : o_ht)), addC);
    }
    // ---- accumulate step EFE ---------------------------------------------
    k_combine<<<launchN(n), 256, 0, stream>>>(
        (float*)(ws + o_tr), (const float*)(ws + o_hp), (const float*)(ws + o_ht),
        (const float*)(ws + o_rC), (const float*)(ws + o_vC), (const float*)(ws + o_ent),
        gpow, (int)n);
    gpow *= 0.99f;
    // swap latent buffers
    size_t tmp = zCur; zCur = zNext; zNext = tmp;
  }
  // ---- mean over trajectories -> out[512] ---------------------------------
  k_final<<<dim3(2), dim3(256), 0, stream>>>((const float*)(ws + o_tr), (float*)d_out);
}
